// TokenRouterMoD_48576080118319
// MI455X (gfx1250) — compile-verified, matrix operand order checked by reading, not run on hardware
//
#include <hip/hip_runtime.h>
#include <hip/hip_bf16.h>

// ---------------- problem constants ----------------
#define BB    4
#define TT    8192
#define DD    1024
#define DFFC  4096
#define KCAP  4096          // T * 0.5
#define MTOT  (BB * KCAP)   // 16384 selected tokens total

// ---------------- WMMA vector types ----------------
typedef __attribute__((ext_vector_type(16))) __bf16 v16bf;
typedef __attribute__((ext_vector_type(8)))  __bf16 v8bf;
typedef __attribute__((ext_vector_type(4)))  __bf16 v4bf;
typedef __attribute__((ext_vector_type(8)))  float  v8f;

// ---------------- GEMM tiling ----------------
#define TILE_M 128
#define TILE_N 128
#define TILE_K 32
#define LDS_STRIDE 40   // 32 + 8 bf16 pad -> 80B row stride, 16B-aligned chunks

// =====================================================================
// 0) f32 [R][C]  ->  bf16 [C][R]   (weights transposed once per launch)
// =====================================================================
__global__ __launch_bounds__(256)
void transpose_f32_to_bf16(const float* __restrict__ in, __bf16* __restrict__ outT,
                           int R, int C) {
    __shared__ float tile[32][33];
    const int cT = blockIdx.x * 32;
    const int rT = blockIdx.y * 32;
    const int tx = threadIdx.x;          // 0..31
    const int ty = threadIdx.y;          // 0..7
#pragma unroll
    for (int i = 0; i < 4; ++i)
        tile[ty + 8 * i][tx] = in[(size_t)(rT + ty + 8 * i) * C + cT + tx];
    __syncthreads();
#pragma unroll
    for (int i = 0; i < 4; ++i)
        outT[(size_t)(cT + ty + 8 * i) * R + rT + tx] = (__bf16)tile[tx][ty + 8 * i];
}

// =====================================================================
// 1) router scores: one wave32 per token, D=1024 dot product
// =====================================================================
__global__ __launch_bounds__(256)
void router_scores(const float* __restrict__ x, const float* __restrict__ wr,
                   float* __restrict__ scores) {
    const int tokens = BB * TT;
    const int gwave = (blockIdx.x * blockDim.x + threadIdx.x) >> 5;
    const int lane  = threadIdx.x & 31;
    if (gwave >= tokens) return;
    const float* xp = x + (size_t)gwave * DD;
    float s = 0.f;
#pragma unroll 8
    for (int j = 0; j < DD / 32; ++j)
        s += xp[lane + 32 * j] * wr[lane + 32 * j];
#pragma unroll
    for (int off = 16; off > 0; off >>= 1)
        s += __shfl_xor(s, off, 32);
    if (lane == 0) scores[gwave] = s;
}

__device__ __forceinline__ unsigned f2key(float f) {
    unsigned u = __float_as_uint(f);
    return (u & 0x80000000u) ? ~u : (u | 0x80000000u);   // order-preserving
}

// =====================================================================
// 2) exact top-k threshold per batch row: 4-pass radix select
//    one block (1024 threads) per row; also zeroes compaction counters
//    state[4*b] = thrKey, [..+1] = nEqTake, [..+2] = cntAbove, [..+3] = cntEq
// =====================================================================
__global__ __launch_bounds__(1024)
void radix_select(const float* __restrict__ scores, unsigned* __restrict__ state) {
    const int b   = blockIdx.x;
    const int tid = threadIdx.x;
    __shared__ unsigned hist[256];
    __shared__ unsigned sBin, sCnt;

    unsigned kr[TT / 1024];
#pragma unroll
    for (int j = 0; j < TT / 1024; ++j)
        kr[j] = f2key(scores[b * TT + tid + 1024 * j]);

    unsigned prefix = 0;
    int kRem = KCAP;
#pragma unroll 1
    for (int p = 0; p < 4; ++p) {
        const int shift = 24 - 8 * p;
        const unsigned mask = p ? (0xFFFFFFFFu << (32 - 8 * p)) : 0u;
        if (tid < 256) hist[tid] = 0;
        __syncthreads();
#pragma unroll
        for (int j = 0; j < TT / 1024; ++j)
            if ((kr[j] & mask) == prefix)
                atomicAdd(&hist[(kr[j] >> shift) & 0xFFu], 1u);
        __syncthreads();
        if (tid == 0) {
            unsigned c = 0; int bin = 0;
            for (int bb = 255; bb >= 0; --bb) {
                if ((int)(c + hist[bb]) >= kRem) { bin = bb; break; }
                c += hist[bb];
            }
            sBin = (unsigned)bin; sCnt = c;
        }
        __syncthreads();
        prefix |= sBin << shift;
        kRem   -= (int)sCnt;
        __syncthreads();
    }
    if (tid == 0) {
        state[4 * b + 0] = prefix;            // exact threshold key
        state[4 * b + 1] = (unsigned)kRem;    // #ties to keep
        state[4 * b + 2] = 0u;                // cntAbove
        state[4 * b + 3] = 0u;                // cntEq
    }
}

// =====================================================================
// 3) compaction: assign unique slots, store idx + sigmoid weight
// =====================================================================
__global__ __launch_bounds__(256)
void compact_topk(const float* __restrict__ scores, unsigned* __restrict__ state,
                  int* __restrict__ idx, float* __restrict__ wgt) {
    const int gid = blockIdx.x * blockDim.x + threadIdx.x;   // B*T
    if (gid >= BB * TT) return;
    const int b = gid / TT;
    const int t = gid - b * TT;
    const float s = scores[gid];
    const unsigned key = f2key(s);
    const unsigned thr = state[4 * b + 0];
    const unsigned nEq = state[4 * b + 1];
    const unsigned nAbove = KCAP - nEq;
    int slot = -1;
    if (key > thr) {
        slot = (int)atomicAdd(&state[4 * b + 2], 1u);
    } else if (key == thr) {
        unsigned p = atomicAdd(&state[4 * b + 3], 1u);
        if (p < nEq) slot = (int)(nAbove + p);
    }
    if (slot >= 0) {
        idx[b * KCAP + slot] = t;
        wgt[b * KCAP + slot] = 1.f / (1.f + expf(-s));
    }
}

// =====================================================================
// 4) gather selected tokens -> bf16 Xsel [MTOT][DD]
// =====================================================================
__global__ __launch_bounds__(256)
void gather_tokens(const float* __restrict__ x, const int* __restrict__ idx,
                   __bf16* __restrict__ Xsel) {
    const size_t gid = (size_t)blockIdx.x * blockDim.x + threadIdx.x; // MTOT*DD/4
    const int m = (int)(gid / (DD / 4));
    const int c = (int)(gid % (DD / 4));
    const int b = m >> 12;                 // m / KCAP
    const int tok = idx[m];
    const float4 v = ((const float4*)x)[((size_t)b * TT + tok) * (DD / 4) + c];
    v4bf o;
    o.x = (__bf16)v.x; o.y = (__bf16)v.y; o.z = (__bf16)v.z; o.w = (__bf16)v.w;
    ((v4bf*)Xsel)[(size_t)m * (DD / 4) + c] = o;
}

// =====================================================================
// 5) out = x  (residual stream base)
// =====================================================================
__global__ __launch_bounds__(256)
void copy_x(const float4* __restrict__ x, float4* __restrict__ out) {
    const size_t gid = (size_t)blockIdx.x * blockDim.x + threadIdx.x;
    out[gid] = x[gid];
}

// =====================================================================
// 6) bf16 WMMA GEMM core, 128x128x32 block tile, double-buffered LDS,
//    CDNA5 async global->LDS staging (GLOBAL_LOAD_ASYNC_TO_LDS_B128,
//    tracked by ASYNCcnt; no VGPR round-trip, overlaps the WMMAs).
//    A: [MTOT][KDIM] bf16 row-major;  Bt: [NDIM][KDIM] bf16 (N-major).
//    EPI=0: Hout = bf16(gelu_exact(acc))             [GEMM1]
//    EPI=1: out[b,idx[m],n] += (acc - Xsel[m,n])*w   [GEMM2, fused scatter]
// =====================================================================
template <int KDIM, int NDIM, int EPI>
__global__ __launch_bounds__(256)
void gemm_bf16_wmma(const __bf16* __restrict__ A, const __bf16* __restrict__ Bt,
                    __bf16* __restrict__ Hout,
                    const int* __restrict__ idx, const float* __restrict__ wgt,
                    const __bf16* __restrict__ Xsel, float* __restrict__ out) {
    __shared__ __align__(16) __bf16 As[2][TILE_M * LDS_STRIDE];
    __shared__ __align__(16) __bf16 Bs[2][TILE_N * LDS_STRIDE];

    const int m0 = blockIdx.y * TILE_M;
    const int n0 = blockIdx.x * TILE_N;

    const int tid  = threadIdx.x;
    const int lane = tid & 31;
    const int wid  = tid >> 5;            // 8 waves
    const int wM   = (wid & 3) * 32;      // 4 waves along M: 32 rows each
    const int wN   = (wid >> 2) * 64;     // 2 waves along N: 64 cols each
    const int half = lane >> 4;
    const int l15  = lane & 15;

    // async global->LDS staging: each thread fills 2x16B of As and 2x16B of Bs
    auto loadTileAsync = [&](int kk, int buf) {
        const int k0 = kk * TILE_K;
#pragma unroll
        for (int i = 0; i < 2; ++i) {
            const int id  = tid + 256 * i;     // 0..511
            const int row = id >> 2;
            const int ch  = (id & 3) * 8;
            // low 32 bits of a generic shared pointer == LDS byte offset
            const unsigned la =
                (unsigned)(uintptr_t)&As[buf][row * LDS_STRIDE + ch];
            const unsigned lb =
                (unsigned)(uintptr_t)&Bs[buf][row * LDS_STRIDE + ch];
            const __bf16* ga = &A [(size_t)(m0 + row) * KDIM + k0 + ch];
            const __bf16* gb = &Bt[(size_t)(n0 + row) * KDIM + k0 + ch];
            asm volatile("global_load_async_to_lds_b128 %0, %1, off"
                         :: "v"(la), "v"(ga) : "memory");
            asm volatile("global_load_async_to_lds_b128 %0, %1, off"
                         :: "v"(lb), "v"(gb) : "memory");
        }
    };
    auto ldfrag = [&](const __bf16* base, int row) -> v16bf {
        v8bf lo = *(const v8bf*)&base[row * LDS_STRIDE + half * 8];
        v8bf hi = *(const v8bf*)&base[row * LDS_STRIDE + 16 + half * 8];
        return __builtin_shufflevector(lo, hi, 0, 1, 2, 3, 4, 5, 6, 7,
                                       8, 9, 10, 11, 12, 13, 14, 15);
    };

    v8f acc[2][4] = {};

    constexpr int nK = KDIM / TILE_K;
    loadTileAsync(0, 0);
    asm volatile("s_wait_asynccnt 0" ::: "memory");
    __syncthreads();
#pragma unroll 1
    for (int kk = 0; kk < nK; ++kk) {
        const int buf = kk & 1;
        if (kk + 1 < nK) loadTileAsync(kk + 1, buf ^ 1);   // overlaps WMMAs

        v16bf af[2], bfv[4];
#pragma unroll
        for (int mt = 0; mt < 2; ++mt) af[mt] = ldfrag(As[buf], wM + mt * 16 + l15);
#pragma unroll
        for (int nt = 0; nt < 4; ++nt) bfv[nt] = ldfrag(Bs[buf], wN + nt * 16 + l15);
#pragma unroll
        for (int mt = 0; mt < 2; ++mt)
#pragma unroll
            for (int nt = 0; nt < 4; ++nt)
                acc[mt][nt] = __builtin_amdgcn_wmma_f32_16x16x32_bf16(
                    false, af[mt], false, bfv[nt], (short)0, acc[mt][nt],
                    false, false);

        if (kk + 1 < nK) asm volatile("s_wait_asynccnt 0" ::: "memory");
        __syncthreads();
    }

    // ---------------- epilogue ----------------
#pragma unroll
    for (int mt = 0; mt < 2; ++mt) {
#pragma unroll
        for (int nt = 0; nt < 4; ++nt) {
            const int mG0 = m0 + wM + mt * 16 + 8 * half;
            const int nG  = n0 + wN + nt * 16 + l15;
            if (EPI == 0) {
#pragma unroll
                for (int r = 0; r < 8; ++r) {
                    const float v = acc[mt][nt][r];
                    const float g = 0.5f * v * (1.0f + erff(v * 0.70710678118654752f));
                    Hout[(size_t)(mG0 + r) * NDIM + nG] = (__bf16)g;
                }
            } else {
#pragma unroll
                for (int r = 0; r < 8; ++r) {
                    const int m   = mG0 + r;
                    const int b   = m >> 12;            // m / KCAP
                    const int tok = idx[m];
                    const float w = wgt[m];
                    const float xs = (float)Xsel[(size_t)m * DD + nG];
                    const float delta = (acc[mt][nt][r] - xs) * w;
                    const size_t o = ((size_t)b * TT + tok) * DD + nG;
                    out[o] += delta;   // unique (tok,n) per lane-elem: no atomics
                }
            }
        }
    }
}

// =====================================================================
// launch
// =====================================================================
extern "C" void kernel_launch(void* const* d_in, const int* in_sizes, int n_in,
                              void* d_out, int out_size, void* d_ws, size_t ws_size,
                              hipStream_t stream) {
    const float* x  = (const float*)d_in[0];
    const float* wr = (const float*)d_in[1];
    const float* W1 = (const float*)d_in[2];   // [DD][DFFC]
    const float* W2 = (const float*)d_in[3];   // [DFFC][DD]
    float* out = (float*)d_out;

    char* ws = (char*)d_ws;
    size_t off = 0;
    auto carve = [&](size_t bytes) -> void* {
        void* p = ws + off;
        off = (off + bytes + 255) & ~(size_t)255;
        return p;
    };
    float*    scores = (float*)   carve((size_t)BB * TT * 4);
    unsigned* state  = (unsigned*)carve((size_t)BB * 4 * 4);
    int*      idx    = (int*)     carve((size_t)MTOT * 4);
    float*    wgt    = (float*)   carve((size_t)MTOT * 4);
    __bf16*   W1t    = (__bf16*)  carve((size_t)DD * DFFC * 2);   // [DFFC][DD]
    __bf16*   W2t    = (__bf16*)  carve((size_t)DFFC * DD * 2);   // [DD][DFFC]
    __bf16*   Xsel   = (__bf16*)  carve((size_t)MTOT * DD * 2);
    __bf16*   H      = (__bf16*)  carve((size_t)MTOT * DFFC * 2);

    // weights -> bf16, transposed to [N][K]
    transpose_f32_to_bf16<<<dim3(DFFC / 32, DD / 32), dim3(32, 8), 0, stream>>>(
        W1, W1t, DD, DFFC);
    transpose_f32_to_bf16<<<dim3(DD / 32, DFFC / 32), dim3(32, 8), 0, stream>>>(
        W2, W2t, DFFC, DD);

    // routing
    router_scores<<<(BB * TT * 32) / 256, 256, 0, stream>>>(x, wr, scores);
    radix_select<<<BB, 1024, 0, stream>>>(scores, state);
    compact_topk<<<(BB * TT) / 256, 256, 0, stream>>>(scores, state, idx, wgt);
    gather_tokens<<<((size_t)MTOT * (DD / 4)) / 256, 256, 0, stream>>>(x, idx, Xsel);

    // residual base
    copy_x<<<((size_t)BB * TT * DD / 4) / 256, 256, 0, stream>>>(
        (const float4*)x, (float4*)out);

    // FFN: H = gelu(Xsel @ W1) ; out[b,idx,:] += (H @ W2 - Xsel) * sigmoid(score)
    gemm_bf16_wmma<DD, DFFC, 0><<<dim3(DFFC / TILE_N, MTOT / TILE_M), 256, 0, stream>>>(
        Xsel, W1t, H, nullptr, nullptr, nullptr, nullptr);
    gemm_bf16_wmma<DFFC, DD, 1><<<dim3(DD / TILE_N, MTOT / TILE_M), 256, 0, stream>>>(
        H, W2t, nullptr, idx, wgt, Xsel, out);
}
